// MultiHeadAttention_73890617360452
// MI455X (gfx1250) — matrix operand order repaired and checked by reading, not compiled, not run on hardware
//
#include <hip/hip_runtime.h>

typedef __attribute__((ext_vector_type(16))) __bf16 v16bf;
typedef __attribute__((ext_vector_type(8)))  float  v8f;

#if __has_builtin(__builtin_amdgcn_global_load_async_to_lds_b64)
#define USE_ASYNC_LDS 1
#else
#define USE_ASYNC_LDS 0
#endif

#if USE_ASYNC_LDS
// Exact parameter types per hipcc diagnostic: pointee = vector_size(8) int,
// global side in AS(1) ("__device__"), LDS side in AS(3) ("__shared__").
typedef int v2i_t __attribute__((vector_size(2 * sizeof(int))));
typedef __attribute__((address_space(1))) v2i_t* as1_v2i;
typedef __attribute__((address_space(3))) v2i_t* as3_v2i;

__device__ __forceinline__ void async_copy_b64(const void* g, void* l) {
  __builtin_amdgcn_global_load_async_to_lds_b64((as1_v2i)g, (as3_v2i)l,
                                                /*imm offset=*/0, /*cpol=*/0);
}
#endif

__device__ __forceinline__ void wait_async0() {
#if USE_ASYNC_LDS
  asm volatile("s_wait_asynccnt 0" ::: "memory");
#endif
}

// ---------------- WMMA helpers (CDNA5 gfx1250, wave32) ----------------

__device__ __forceinline__ v8f wmma_bf16(v16bf a, v16bf b, v8f c) {
  // v_wmma_f32_16x16x32_bf16  D = A(16x32) * B(32x16) + C
  return __builtin_amdgcn_wmma_f32_16x16x32_bf16(
      /*neg_a=*/false, a, /*neg_b=*/false, b,
      /*c_mod=*/(short)0, c, /*reuse_a=*/false, /*reuse_b=*/false);
}

// A fragment: 16x32 bf16, rows m = lane&15; half (lane>>4) selects K groups
// VGPR0..3 : K = 8h .. 8h+7 ; VGPR4..7 : K = 16+8h .. 16+8h+7  (ISA 7.12.2)
__device__ __forceinline__ v16bf frag_a(const __bf16* tile, int pitch, int koff, int lane) {
  const int h = (lane >> 4) & 1;
  const int m = lane & 15;
  const __bf16* p0 = tile + m * pitch + koff + 8 * h;
  const __bf16* p1 = p0 + 16;
  v16bf f;
#pragma unroll
  for (int i = 0; i < 8; ++i) f[i] = p0[i];
#pragma unroll
  for (int i = 0; i < 8; ++i) f[8 + i] = p1[i];
  return f;
}

// B fragment: 32x16 bf16 staged n-major (Bs[n][k]); lane = column n (lane&15),
// half selects K = 8h .. 8h+7 and 16+8h .. 16+8h+7 (same K order as A fragment)
__device__ __forceinline__ v16bf frag_b(const __bf16* tile, int pitch, int koff, int lane) {
  const int h = (lane >> 4) & 1;
  const int n = lane & 15;
  const __bf16* p = tile + n * pitch + koff + 8 * h;
  v16bf f;
#pragma unroll
  for (int i = 0; i < 8; ++i) f[i] = p[i];
#pragma unroll
  for (int i = 0; i < 8; ++i) f[8 + i] = p[16 + i];
  return f;
}

// ---------------- GEMM + bias: out[4096x1024] = A[4096x1024] @ W[1024x1024] + b ----
// Tile 128x128x32, 256 threads = 8 waves (4 M-waves x 2 N-waves),
// each wave: 2x4 = 8 wmma tiles of 16x16. Double-buffered LDS (ping-pong),
// one barrier per K-step: next tile's global loads overlap current WMMAs.

template <bool A_IS_BF16, bool OUT_BF16>
__global__ __launch_bounds__(256) void gemm_bias_k(const void* __restrict__ Ain,
                                                   const float* __restrict__ W,
                                                   const float* __restrict__ bias,
                                                   void* __restrict__ Out) {
  constexpr int Kdim = 1024;
  constexpr int LD = 1024;
  __shared__ __bf16 As[2][128 * 40];  // [m][k], pitch 40 (80B, 16B aligned)
  __shared__ __bf16 Bs[2][128 * 40];  // [n][k] (W transposed), pitch 40

  const int tid = threadIdx.x;
  const int lane = tid & 31;
  const int w = tid >> 5;
  const int wm = w & 3;   // 0..3 -> 32 rows each
  const int wn = w >> 2;  // 0..1 -> 64 cols each
  const int n0 = blockIdx.x * 128;
  const int m0 = blockIdx.y * 128;

  v8f acc[2][4] = {};

  auto stageA = [&](int k0, int p) {
    if constexpr (A_IS_BF16) {
      const __bf16* A = (const __bf16*)Ain;
#pragma unroll
      for (int i = 0; i < 4; ++i) {
        int c = tid + 256 * i;          // 1024 chunks of 4 bf16
        int row = c >> 3, c4 = c & 7;
        uint2 v = *reinterpret_cast<const uint2*>(A + (size_t)(m0 + row) * LD + k0 + c4 * 4);
        *reinterpret_cast<uint2*>(&As[p][row * 40 + c4 * 4]) = v;
      }
    } else {
      const float* A = (const float*)Ain;
#pragma unroll
      for (int i = 0; i < 4; ++i) {
        int c = tid + 256 * i;
        int row = c >> 3, c4 = c & 7;
        const float4 v = *reinterpret_cast<const float4*>(A + (size_t)(m0 + row) * LD + k0 + c4 * 4);
        __bf16 t[4] = {(__bf16)v.x, (__bf16)v.y, (__bf16)v.z, (__bf16)v.w};
        *reinterpret_cast<uint2*>(&As[p][row * 40 + c4 * 4]) = *reinterpret_cast<uint2*>(t);
      }
    }
  };
  auto stageB = [&](int k0, int p) {
#pragma unroll
    for (int i = 0; i < 4; ++i) {
      int c = tid + 256 * i;            // 1024 chunks of 4 floats
      int kk = c >> 5, nc4 = c & 31;
      const float4 v = *reinterpret_cast<const float4*>(W + (size_t)(k0 + kk) * LD + n0 + nc4 * 4);
      Bs[p][(nc4 * 4 + 0) * 40 + kk] = (__bf16)v.x;
      Bs[p][(nc4 * 4 + 1) * 40 + kk] = (__bf16)v.y;
      Bs[p][(nc4 * 4 + 2) * 40 + kk] = (__bf16)v.z;
      Bs[p][(nc4 * 4 + 3) * 40 + kk] = (__bf16)v.w;
    }
  };

  stageA(0, 0);
  stageB(0, 0);
  int p = 0;
  for (int k0 = 0; k0 < Kdim; k0 += 32) {
    __syncthreads();  // buffer p fully staged; buffer p^1 free
    if (k0 + 32 < Kdim) {
      __builtin_prefetch((const char*)W + ((size_t)(k0 + 64) * LD + n0) * 4, 0, 1);
      stageA(k0 + 32, p ^ 1);
      stageB(k0 + 32, p ^ 1);
    }
#pragma unroll
    for (int mi = 0; mi < 2; ++mi) {
      v16bf a = frag_a(&As[p][(wm * 32 + mi * 16) * 40], 40, 0, lane);
#pragma unroll
      for (int ni = 0; ni < 4; ++ni) {
        v16bf b = frag_b(&Bs[p][(wn * 64 + ni * 16) * 40], 40, 0, lane);
        acc[mi][ni] = wmma_bf16(a, b, acc[mi][ni]);
      }
    }
    p ^= 1;
  }

  // ---- epilogue: bias add, store. C/D layout: elem r -> (m = r + 8*half, n = lane&15)
  const int h = (lane >> 4) & 1;
  const int nl = lane & 15;
  float bv[4];
#pragma unroll
  for (int ni = 0; ni < 4; ++ni) bv[ni] = bias[n0 + wn * 64 + ni * 16 + nl];
#pragma unroll
  for (int mi = 0; mi < 2; ++mi) {
#pragma unroll
    for (int ni = 0; ni < 4; ++ni) {
      const int ng = n0 + wn * 64 + ni * 16 + nl;
#pragma unroll
      for (int r = 0; r < 8; ++r) {
        const int mg = m0 + wm * 32 + mi * 16 + r + 8 * h;
        float v = acc[mi][ni][r] + bv[ni];
        if constexpr (OUT_BF16)
          ((__bf16*)Out)[(size_t)mg * LD + ng] = (__bf16)v;
        else
          ((float*)Out)[(size_t)mg * LD + ng] = v;
      }
    }
  }
}

// ---------------- Flash attention: per (b,h), 64-query tile, stream KV in 64-chunks.
// 128 threads = 4 waves; wave w owns query rows 16w..16w+15 and the full DH=64 output.
// Q/K tiles staged via GLOBAL_LOAD_ASYNC_TO_LDS (ASYNCcnt) when available.

__global__ __launch_bounds__(128) void attn_k(const __bf16* __restrict__ Q,
                                              const __bf16* __restrict__ Kb,
                                              const __bf16* __restrict__ Vb,
                                              __bf16* __restrict__ O) {
  constexpr int S = 2048, Dm = 1024, DH = 64;
  __shared__ __bf16 Qs[64 * 72];    // [q][dh]
  __shared__ __bf16 Ks[64 * 72];    // [kv][dh]  (== Bs[n][k] for Q@K^T)
  __shared__ __bf16 Vts[64 * 72];   // [dh][kv]  (== Bs[n][k] for P@V)
  __shared__ __bf16 Ps[64 * 72];    // [q][kv] probabilities (A for P@V)
  __shared__ float Ssc[64 * 68];    // raw scores f32
  __shared__ float pmax[64 * 4], psum[64 * 4];
  __shared__ float rmax[64], rsum[64], alphaS[64];

  const int tid = threadIdx.x, lane = tid & 31, w = tid >> 5;
  const int bh = blockIdx.y, b = bh >> 4, hh = bh & 15;
  const int q0 = blockIdx.x * 64;
  const int h = (lane >> 4) & 1, nl = lane & 15;

  const __bf16* Qg = Q + ((size_t)(b * S + q0)) * Dm + hh * DH;
#pragma unroll
  for (int i = 0; i < 8; ++i) {
    int c = tid + 128 * i;            // 1024 chunks of 4 bf16
    int row = c >> 4, c4 = (c & 15) * 4;
#if USE_ASYNC_LDS
    async_copy_b64(Qg + (size_t)row * Dm + c4, &Qs[row * 72 + c4]);
#else
    *reinterpret_cast<uint2*>(&Qs[row * 72 + c4]) =
        *reinterpret_cast<const uint2*>(Qg + (size_t)row * Dm + c4);
#endif
  }
  if (tid < 64) { rmax[tid] = -3.0e38f; rsum[tid] = 0.f; }
  v8f o[4] = {};

  for (int kv0 = 0; kv0 < S; kv0 += 64) {
    __syncthreads();  // previous iteration's Ps/Vts consumers done
    const __bf16* Kg = Kb + ((size_t)(b * S + kv0)) * Dm + hh * DH;
    const __bf16* Vg = Vb + ((size_t)(b * S + kv0)) * Dm + hh * DH;
#pragma unroll
    for (int i = 0; i < 8; ++i) {
      int c = tid + 128 * i;
      int row = c >> 4, c4 = (c & 15) * 4;
#if USE_ASYNC_LDS
      async_copy_b64(Kg + (size_t)row * Dm + c4, &Ks[row * 72 + c4]);
#else
      *reinterpret_cast<uint2*>(&Ks[row * 72 + c4]) =
          *reinterpret_cast<const uint2*>(Kg + (size_t)row * Dm + c4);
#endif
      union { uint2 u; __bf16 e[4]; } uv;
      uv.u = *reinterpret_cast<const uint2*>(Vg + (size_t)row * Dm + c4);
#pragma unroll
      for (int j = 0; j < 4; ++j) Vts[(c4 + j) * 72 + row] = uv.e[j];
    }
    wait_async0();     // ASYNCcnt -> 0 before the workgroup barrier
    __syncthreads();

    // scores S = (Q K^T) * 1/sqrt(64): wave w -> rows w*16.., 4 n-tiles, 2 k-steps
#pragma unroll
    for (int ni = 0; ni < 4; ++ni) {
      v8f s = {};
#pragma unroll
      for (int ks = 0; ks < 2; ++ks) {
        v16bf a = frag_a(&Qs[w * 16 * 72], 72, ks * 32, lane);
        v16bf bb = frag_b(&Ks[ni * 16 * 72], 72, ks * 32, lane);
        s = wmma_bf16(a, bb, s);
      }
#pragma unroll
      for (int r = 0; r < 8; ++r)
        Ssc[(w * 16 + r + 8 * h) * 68 + ni * 16 + nl] = s[r] * 0.125f;
    }
    __syncthreads();

    // online softmax: partial row max (256 slots = 64 rows x 4 quarters)
#pragma unroll
    for (int s2 = 0; s2 < 2; ++s2) {
      int slot = tid + 128 * s2;
      int row = slot >> 2, qq = slot & 3;
      float m = -3.0e38f;
#pragma unroll
      for (int j = 0; j < 16; ++j) m = fmaxf(m, Ssc[row * 68 + qq * 16 + j]);
      pmax[row * 4 + qq] = m;
    }
    __syncthreads();
    if (tid < 64) {
      float mn = rmax[tid];
#pragma unroll
      for (int qq = 0; qq < 4; ++qq) mn = fmaxf(mn, pmax[tid * 4 + qq]);
      alphaS[tid] = __expf(rmax[tid] - mn);
      rmax[tid] = mn;
    }
    __syncthreads();
    // p = exp(s - max), pack to bf16, partial sums
#pragma unroll
    for (int s2 = 0; s2 < 2; ++s2) {
      int slot = tid + 128 * s2;
      int row = slot >> 2, qq = slot & 3;
      float mn = rmax[row];
      float sum = 0.f;
#pragma unroll
      for (int j = 0; j < 16; ++j) {
        float pv = __expf(Ssc[row * 68 + qq * 16 + j] - mn);
        Ps[row * 72 + qq * 16 + j] = (__bf16)pv;
        sum += pv;
      }
      psum[row * 4 + qq] = sum;
    }
    __syncthreads();
    if (tid < 64)
      rsum[tid] = alphaS[tid] * rsum[tid] +
                  psum[tid * 4] + psum[tid * 4 + 1] + psum[tid * 4 + 2] + psum[tid * 4 + 3];

    // rescale running output, then O += P @ V
    float av[8];
#pragma unroll
    for (int r = 0; r < 8; ++r) av[r] = alphaS[w * 16 + r + 8 * h];
#pragma unroll
    for (int ni = 0; ni < 4; ++ni)
#pragma unroll
      for (int r = 0; r < 8; ++r) o[ni][r] *= av[r];
#pragma unroll
    for (int ks = 0; ks < 2; ++ks) {
      v16bf a = frag_a(&Ps[w * 16 * 72], 72, ks * 32, lane);
#pragma unroll
      for (int ni = 0; ni < 4; ++ni) {
        v16bf bb = frag_b(&Vts[ni * 16 * 72], 72, ks * 32, lane);
        o[ni] = wmma_bf16(a, bb, o[ni]);
      }
    }
  }
  __syncthreads();

  float inv[8];
#pragma unroll
  for (int r = 0; r < 8; ++r) inv[r] = 1.f / rsum[w * 16 + r + 8 * h];
  __bf16* Og = O + ((size_t)(b * S + q0)) * Dm + hh * DH;
#pragma unroll
  for (int ni = 0; ni < 4; ++ni)
#pragma unroll
    for (int r = 0; r < 8; ++r)
      Og[(size_t)(w * 16 + r + 8 * h) * Dm + ni * 16 + nl] = (__bf16)(o[ni][r] * inv[r]);
}

// ---------------- launcher ----------------

extern "C" void kernel_launch(void* const* d_in, const int* in_sizes, int n_in,
                              void* d_out, int out_size, void* d_ws, size_t ws_size,
                              hipStream_t stream) {
  (void)in_sizes; (void)n_in; (void)out_size; (void)ws_size;
  const float* query = (const float*)d_in[0];
  const float* key   = (const float*)d_in[1];
  const float* value = (const float*)d_in[2];
  const float* Wq = (const float*)d_in[3];  const float* bq = (const float*)d_in[4];
  const float* Wk = (const float*)d_in[5];  const float* bk = (const float*)d_in[6];
  const float* Wv = (const float*)d_in[7];  const float* bv = (const float*)d_in[8];
  const float* Wo = (const float*)d_in[9];  const float* bo = (const float*)d_in[10];

  constexpr size_t NACT = 4096ull * 1024ull;  // B*S x D elements
  __bf16* Qb = (__bf16*)d_ws;
  __bf16* Kb = Qb + NACT;
  __bf16* Vb = Kb + NACT;
  __bf16* Ab = Vb + NACT;   // concat-heads attention output (bf16)

  dim3 gg(8, 32);           // N/128, M/128
  gemm_bias_k<false, true><<<gg, 256, 0, stream>>>(query, Wq, bq, Qb);
  gemm_bias_k<false, true><<<gg, 256, 0, stream>>>(key,   Wk, bk, Kb);
  gemm_bias_k<false, true><<<gg, 256, 0, stream>>>(value, Wv, bv, Vb);
  attn_k<<<dim3(32, 32), 128, 0, stream>>>(Qb, Kb, Vb, Ab);
  gemm_bias_k<true, false><<<gg, 256, 0, stream>>>(Ab, Wo, bo, d_out);
}
